// FFM_64811056496790
// MI455X (gfx1250) — matrix operand order repaired and checked by reading, not compiled, not run on hardware
//
#include <hip/hip_runtime.h>

// ---------------------------------------------------------------------------
// FFM forward for MI455X (gfx1250, wave32, WMMA).
//   GEMMs: bf16 WMMA (v_wmma_f32_16x16x32_bf16), fp32 accumulate; A and B
//          both staged with double-buffered GLOBAL_LOAD_ASYNC_TO_LDS_B128
//          (ASYNCcnt-tracked).  x is pre-converted to bf16 once so no
//          conversion code lives in any GEMM inner loop.
//   Scan:  chunked parallel linear recurrence, c-dimension == 32 lanes.
// ---------------------------------------------------------------------------

#define T_DIM   8192
#define IN_DIM  1024
#define OUT_DIM 1024
#define M_DIM   128
#define C_DIM   32
#define NCH     32            // chunks in the parallel scan
#define LCH     (T_DIM / NCH) // 256 steps per chunk

typedef __attribute__((ext_vector_type(16))) __bf16 v16bf;
typedef __attribute__((ext_vector_type(8)))  float  v8f;
typedef __attribute__((ext_vector_type(4)))  int    v4i_t;

struct BF16Frag { uint4 lo; uint4 hi; };   // 32 bytes == v16bf

#define AS1 __attribute__((address_space(1)))
#define AS3 __attribute__((address_space(3)))

#if __has_builtin(__builtin_amdgcn_global_load_async_to_lds_b128)
#define HAVE_ASYNC_LDS 1
#endif

// ---- fp32 -> bf16 (native cast; backend picks v_cvt_pk_bf16_f32 if avail) -
__device__ __forceinline__ unsigned short f2bfbits(float f) {
  __bf16 h = (__bf16)f;
  return __builtin_bit_cast(unsigned short, h);
}
__device__ __forceinline__ unsigned pk2bf(float x, float y) {
  return (unsigned)f2bfbits(x) | ((unsigned)f2bfbits(y) << 16);
}

// ---- async memory -> LDS copy (16B per lane) -----------------------------
// builtin signature (probe-confirmed): (int4 AS1*, int4 AS3*, Ii off, Ii cpol)
__device__ __forceinline__ void async_cp16(const void* g, void* l) {
#ifdef HAVE_ASYNC_LDS
  void* gnc = const_cast<void*>(g);
  __builtin_amdgcn_global_load_async_to_lds_b128(
      (AS1 v4i_t*)gnc, (AS3 v4i_t*)l, 0, 0);
#else
  *(uint4*)l = *(const uint4*)g;        // synchronous fallback
#endif
}

__device__ __forceinline__ void wait_async_le3(void) {
#ifdef HAVE_ASYNC_LDS
#if __has_builtin(__builtin_amdgcn_s_wait_asynccnt)
  __builtin_amdgcn_s_wait_asynccnt(3);
#else
  asm volatile("s_wait_asynccnt 0x3" ::: "memory");
#endif
#endif
}
__device__ __forceinline__ void wait_async_0(void) {
#ifdef HAVE_ASYNC_LDS
#if __has_builtin(__builtin_amdgcn_s_wait_asynccnt)
  __builtin_amdgcn_s_wait_asynccnt(0);
#else
  asm volatile("s_wait_asynccnt 0x0" ::: "memory");
#endif
#endif
}

// ---------------------------------------------------------------------------
// fp32 -> bf16 streaming convert (weights once per launch, and x)
// ---------------------------------------------------------------------------
__global__ __launch_bounds__(256) void cvt_f32_bf16(const float* __restrict__ s,
                                                    unsigned* __restrict__ d,
                                                    long npairs) {
  long i = (long)blockIdx.x * blockDim.x + threadIdx.x;
  long stride = (long)gridDim.x * blockDim.x;
  for (; i < npairs; i += stride) {
    const float2 v = *(const float2*)(s + i * 2);
    d[i] = pk2bf(v.x, v.y);
  }
}

// ---------------------------------------------------------------------------
// GEMM: C[rows x N] = A[rows x K] * B[N x K]^T + bias   (NT, both K-major)
// A, B raw-bf16 in global memory; both staged via async-to-LDS.
// Block tile 64x128, 8 waves (4 along M x 2 along N), each wave 16x64 ->
// 4x v_wmma_f32_16x16x32_bf16 per 32-wide k-step.  Double-buffered LDS:
// tile k+1's async loads (3 b128/thread) overlap tile k's WMMA work;
// in-order ASYNCcnt completion means asynccnt<=3 implies tile k landed.
// Fragment layouts per ISA 7.12.2 (wave32):
//   A 16x32 bf16: lane L row M=L%16; lanes<16 K={0..7,16..23},
//                 lanes>=16 K={8..15,24..31}  -> two ds_read_b128
//   B 32x16 bf16: lane L col N=L%16; lanes<16 K={0..7,16..23},
//                 lanes>=16 K={8..15,24..31}
//   C/D f32:      lane L col N=L%16; VGPR r -> row M = r + (L>=16 ? 8 : 0)
// ---------------------------------------------------------------------------
__global__ __launch_bounds__(256, 2)
void gemm_nt_wmma(const unsigned short* __restrict__ A,
                  const unsigned short* __restrict__ B,
                  const float* __restrict__ bias,
                  float* __restrict__ Cout,
                  int K, int N) {
  __shared__ alignas(16) unsigned short As[2][64 * 32];   // 2 x 4 KB
  __shared__ alignas(16) unsigned short Bs[2][128 * 32];  // 2 x 8 KB

  const int tid  = threadIdx.x;
  const int lane = tid & 31;
  const int w    = tid >> 5;
  const int wm   = w & 3;        // wave row-group (16 rows each)
  const int wn   = w >> 2;       // wave col-group (64 cols each)
  const long rowBase = (long)blockIdx.x * 64;
  const int  colBase = blockIdx.y * 128;

  v8f acc[4] = {};
  const int nk = K >> 5;

  // stage tile kt into LDS buffer pb: 1 async b128 for A + 2 for B per thread
  auto issueTile = [&](int kt, int pb) {
    const int k0 = kt << 5;
    {
      const int r = tid >> 2, ko = (tid & 3) << 3;
      async_cp16(A + (rowBase + r) * (long)K + k0 + ko,
                 &As[pb][r * 32 + ko]);
    }
#pragma unroll
    for (int i = 0; i < 2; ++i) {
      const int e = tid + (i << 8);             // 512 uint4 chunks
      const int r = e >> 2, ko = (e & 3) << 3;
      async_cp16(B + (long)(colBase + r) * K + k0 + ko,
                 &Bs[pb][r * 32 + ko]);
    }
  };

  issueTile(0, 0);

  for (int kt = 0; kt < nk; ++kt) {
    const int p = kt & 1;
    const bool more = (kt + 1 < nk);
    if (more) issueTile(kt + 1, 1 - p);

    // tile kt's async loads (issued one iteration ago) are complete once
    // asynccnt drops to the 3 just issued for tile kt+1 (in-order).
    if (more) wait_async_le3();
    else      wait_async_0();
    __syncthreads();

    const unsigned short* Ab = As[p];
    const unsigned short* Bb = Bs[p];

    // ---- A fragment for this wave's 16 rows ------------------------------
    const int am  = (lane & 15) + (wm << 4);
    const int akb = (lane < 16) ? 0 : 8;
    BF16Frag af;
    af.lo = *(const uint4*)(Ab + am * 32 + akb);        // K akb..akb+7
    af.hi = *(const uint4*)(Ab + am * 32 + akb + 16);   // K akb+16..akb+23
    const v16bf av = __builtin_bit_cast(v16bf, af);

    // ---- 4 B fragments + WMMA --------------------------------------------
#pragma unroll
    for (int j = 0; j < 4; ++j) {
      const int bn  = (wn << 6) + (j << 4) + (lane & 15);
      const int bkb = (lane < 16) ? 0 : 8;
      BF16Frag bf;
      bf.lo = *(const uint4*)(Bb + bn * 32 + bkb);
      bf.hi = *(const uint4*)(Bb + bn * 32 + bkb + 16);
      const v16bf bv = __builtin_bit_cast(v16bf, bf);
      acc[j] = __builtin_amdgcn_wmma_f32_16x16x32_bf16(
          false, av, false, bv, (short)0, acc[j], false, false);
    }
    __syncthreads();   // all waves done reading buf p before it is re-staged
  }

  // ---- epilogue: bias + store --------------------------------------------
  const long mrow0 = rowBase + (wm << 4) + ((lane >= 16) ? 8 : 0);
#pragma unroll
  for (int j = 0; j < 4; ++j) {
    const int gn = colBase + (wn << 6) + (j << 4) + (lane & 15);
    const float bv = bias[gn];
#pragma unroll
    for (int r = 0; r < 8; ++r)
      Cout[(mrow0 + r) * (long)N + gn] = acc[j][r] + bv;
  }
}

// ---------------------------------------------------------------------------
// gated_x = (x@pre^T + pre_b) * sigmoid(x@gin^T + gin_b)
// ---------------------------------------------------------------------------
__global__ __launch_bounds__(256) void gate_combine(const float* __restrict__ pre,
                                                    const float* __restrict__ gin,
                                                    float* __restrict__ gated,
                                                    long n) {
  long i = (long)blockIdx.x * blockDim.x + threadIdx.x;
  long stride = (long)gridDim.x * blockDim.x;
  for (; i < n; i += stride)
    gated[i] = pre[i] * (1.f / (1.f + expf(-gin[i])));
}

// ---------------------------------------------------------------------------
// Scan phase 1: per (m, chunk) local scan from zero state; lanes = c (32).
// ---------------------------------------------------------------------------
__global__ __launch_bounds__(32) void scan_phase1(const float* __restrict__ gated,
                                                  const unsigned char* __restrict__ done,
                                                  const float* __restrict__ a,
                                                  const float* __restrict__ b,
                                                  float* __restrict__ bsumRe,
                                                  float* __restrict__ bsumIm,
                                                  int* __restrict__ anydone) {
  const int ch = blockIdx.x, m = blockIdx.y, c = threadIdx.x;
  const float mag = expf(-fabsf(a[m]));
  const float bc = b[c];
  const float gr = mag * cosf(bc), gi = mag * sinf(bc);
  float sr = 0.f, si = 0.f;
  int anyd = 0;
  const int t0 = ch * LCH;
  for (int t = t0; t < t0 + LCH; ++t) {
    const int d = done[t];
    anyd |= d;
    const float keep = d ? 0.f : 1.f;
    const float gx = gated[(long)t * M_DIM + m];
    const float nr = keep * (sr * gr - si * gi) + gx;
    const float ni = keep * (sr * gi + si * gr);
    sr = nr; si = ni;
  }
  const long o = ((long)ch * M_DIM + m) * C_DIM + c;
  bsumRe[o] = sr; bsumIm[o] = si;
  if (m == 0 && c == 0) anydone[ch] = anyd;
}

// ---------------------------------------------------------------------------
// Scan phase 2: serial over 32 chunks, parallel over 4096 (m,c) lanes.
// s_in[ch+1] = A_ch * s_in[ch] + b_ch,  A_ch = anydone ? 0 : gamma^LCH
// ---------------------------------------------------------------------------
__global__ __launch_bounds__(256) void scan_phase2(const float* __restrict__ bsumRe,
                                                   const float* __restrict__ bsumIm,
                                                   const int* __restrict__ anydone,
                                                   const float* __restrict__ state_re,
                                                   const float* __restrict__ state_im,
                                                   const float* __restrict__ a,
                                                   const float* __restrict__ b,
                                                   float* __restrict__ carryRe,
                                                   float* __restrict__ carryIm) {
  const int idx = blockIdx.x * 256 + threadIdx.x;   // 0..4095
  const int m = idx / C_DIM, c = idx % C_DIM;
  float sr = state_re[idx], si = state_im[idx];
  const float magL = expf(-fabsf(a[m]) * (float)LCH);
  const float ang  = b[c] * (float)LCH;
  const float AR = magL * cosf(ang), AI = magL * sinf(ang);
  for (int ch = 0; ch < NCH; ++ch) {
    const long o = ((long)ch * M_DIM + m) * C_DIM + c;
    carryRe[o] = sr; carryIm[o] = si;
    const float cr = anydone[ch] ? 0.f : AR;
    const float ci = anydone[ch] ? 0.f : AI;
    const float nr = sr * cr - si * ci + bsumRe[o];
    const float ni = sr * ci + si * cr + bsumIm[o];
    sr = nr; si = ni;
  }
}

// ---------------------------------------------------------------------------
// Scan phase 3: replay with true carry-in; emit fp32 interleaved complex
// `states` (268 MB -> bandwidth bound) plus a bf16 z_in copy for the mix GEMM.
// ---------------------------------------------------------------------------
__global__ __launch_bounds__(32) void scan_phase3(const float* __restrict__ gated,
                                                  const unsigned char* __restrict__ done,
                                                  const float* __restrict__ a,
                                                  const float* __restrict__ b,
                                                  const float* __restrict__ carryRe,
                                                  const float* __restrict__ carryIm,
                                                  float* __restrict__ statesOut,
                                                  unsigned short* __restrict__ zbf) {
  const int ch = blockIdx.x, m = blockIdx.y, c = threadIdx.x;
  const float mag = expf(-fabsf(a[m]));
  const float bc = b[c];
  const float gr = mag * cosf(bc), gi = mag * sinf(bc);
  const long co = ((long)ch * M_DIM + m) * C_DIM + c;
  float sr = carryRe[co], si = carryIm[co];
  const int t0 = ch * LCH;
  for (int t = t0; t < t0 + LCH; ++t) {
    const float keep = done[t] ? 0.f : 1.f;
    const float gx = gated[(long)t * M_DIM + m];
    const float nr = keep * (sr * gr - si * gi) + gx;
    const float ni = keep * (sr * gi + si * gr);
    sr = nr; si = ni;
    const long so = ((long)t * M_DIM + m) * C_DIM + c;
    *(float2*)(statesOut + so * 2) = make_float2(sr, si);   // complex64 layout
    const long zo = (long)t * (2 * M_DIM * C_DIM) + m * C_DIM + c;
    zbf[zo] = f2bfbits(sr);
    zbf[zo + M_DIM * C_DIM] = f2bfbits(si);
  }
}

// ---------------------------------------------------------------------------
// Fused epilogue: zg = z*sigmoid(gout_logits); LayerNorm(zg); + skip*(1-g)
// ---------------------------------------------------------------------------
__global__ __launch_bounds__(256) void ln_combine(const float* __restrict__ z,
                                                  const float* __restrict__ gl,
                                                  const float* __restrict__ sk,
                                                  float* __restrict__ out) {
  __shared__ float2 red[256];
  const int t = blockIdx.x, tid = threadIdx.x;
  const long base = (long)t * OUT_DIM;
  float zg[4], go[4], sv[4];
  float sum = 0.f, sq = 0.f;
#pragma unroll
  for (int i = 0; i < 4; ++i) {
    const int cx = tid + (i << 8);
    const float g = 1.f / (1.f + expf(-gl[base + cx]));
    const float v = z[base + cx] * g;
    zg[i] = v; go[i] = g; sv[i] = sk[base + cx];
    sum += v; sq += v * v;
  }
  red[tid] = make_float2(sum, sq);
  __syncthreads();
  for (int s = 128; s > 0; s >>= 1) {
    if (tid < s) {
      red[tid].x += red[tid + s].x;
      red[tid].y += red[tid + s].y;
    }
    __syncthreads();
  }
  const float mu  = red[0].x * (1.f / OUT_DIM);
  const float var = red[0].y * (1.f / OUT_DIM) - mu * mu;
  const float inv = rsqrtf(var + 1e-5f);
#pragma unroll
  for (int i = 0; i < 4; ++i) {
    const int cx = tid + (i << 8);
    out[base + cx] = (zg[i] - mu) * inv + sv[i] * (1.f - go[i]);
  }
}

// ---------------------------------------------------------------------------
extern "C" void kernel_launch(void* const* d_in, const int* in_sizes, int n_in,
                              void* d_out, int out_size, void* d_ws, size_t ws_size,
                              hipStream_t stream) {
  (void)in_sizes; (void)n_in; (void)out_size; (void)ws_size;

  const float*         x        = (const float*)d_in[0];
  const unsigned char* done     = (const unsigned char*)d_in[1];  // jax bool -> 1B
  const float*         state_re = (const float*)d_in[2];
  const float*         state_im = (const float*)d_in[3];
  const float*         a        = (const float*)d_in[4];
  const float*         b        = (const float*)d_in[5];
  const float *pre_w = (const float*)d_in[6],  *pre_b = (const float*)d_in[7];
  const float *gin_w = (const float*)d_in[8],  *gin_b = (const float*)d_in[9];
  const float *gout_w = (const float*)d_in[10], *gout_b = (const float*)d_in[11];
  const float *skip_w = (const float*)d_in[12], *skip_b = (const float*)d_in[13];
  const float *mix_w  = (const float*)d_in[14], *mix_b  = (const float*)d_in[15];

  float* outMain   = (float*)d_out;                                   // [T, OUT]
  float* statesOut = outMain + (size_t)T_DIM * OUT_DIM;               // [T,M,C] cplx

  // ---- workspace carve-up (256B aligned) ---------------------------------
  char* wsp = (char*)d_ws;
  size_t off = 0;
  auto alloc = [&](size_t bytes) -> char* {
    char* p = wsp + off;
    off = (off + bytes + 255) & ~(size_t)255;
    return p;
  };
  unsigned short* xbf   = (unsigned short*)alloc((size_t)T_DIM * IN_DIM * 2);
  unsigned short* wPre  = (unsigned short*)alloc((size_t)M_DIM * IN_DIM * 2);
  unsigned short* wGin  = (unsigned short*)alloc((size_t)M_DIM * IN_DIM * 2);
  unsigned short* wGout = (unsigned short*)alloc((size_t)OUT_DIM * IN_DIM * 2);
  unsigned short* wSkip = (unsigned short*)alloc((size_t)OUT_DIM * IN_DIM * 2);
  unsigned short* wMix  = (unsigned short*)alloc((size_t)OUT_DIM * 2 * M_DIM * C_DIM * 2);
  float* tmpPre  = (float*)alloc((size_t)T_DIM * M_DIM * 4);
  float* tmpGin  = (float*)alloc((size_t)T_DIM * M_DIM * 4);
  float* gated   = (float*)alloc((size_t)T_DIM * M_DIM * 4);
  float* goutLog = (float*)alloc((size_t)T_DIM * OUT_DIM * 4);
  float* skipBuf = (float*)alloc((size_t)T_DIM * OUT_DIM * 4);
  float* zBuf    = (float*)alloc((size_t)T_DIM * OUT_DIM * 4);
  unsigned short* zbf = (unsigned short*)alloc((size_t)T_DIM * 2 * M_DIM * C_DIM * 2);
  float* bsumRe  = (float*)alloc((size_t)NCH * M_DIM * C_DIM * 4);
  float* bsumIm  = (float*)alloc((size_t)NCH * M_DIM * C_DIM * 4);
  float* carryRe = (float*)alloc((size_t)NCH * M_DIM * C_DIM * 4);
  float* carryIm = (float*)alloc((size_t)NCH * M_DIM * C_DIM * 4);
  int*   anyd    = (int*)alloc(NCH * 4);

  // ---- 1) x and weights -> bf16 ------------------------------------------
  cvt_f32_bf16<<<4096, 256, 0, stream>>>(x, (unsigned*)xbf, (long)T_DIM * IN_DIM / 2);
  cvt_f32_bf16<<<256, 256, 0, stream>>>(pre_w,  (unsigned*)wPre,  (long)M_DIM * IN_DIM / 2);
  cvt_f32_bf16<<<256, 256, 0, stream>>>(gin_w,  (unsigned*)wGin,  (long)M_DIM * IN_DIM / 2);
  cvt_f32_bf16<<<1024, 256, 0, stream>>>(gout_w, (unsigned*)wGout, (long)OUT_DIM * IN_DIM / 2);
  cvt_f32_bf16<<<1024, 256, 0, stream>>>(skip_w, (unsigned*)wSkip, (long)OUT_DIM * IN_DIM / 2);
  cvt_f32_bf16<<<2048, 256, 0, stream>>>(mix_w,  (unsigned*)wMix,
                                         (long)OUT_DIM * 2 * M_DIM * C_DIM / 2);

  // ---- 2) input gating GEMMs + merge -------------------------------------
  dim3 gSmall(T_DIM / 64, M_DIM / 128);       // (128, 1)
  gemm_nt_wmma<<<gSmall, 256, 0, stream>>>(xbf, wPre, pre_b, tmpPre, IN_DIM, M_DIM);
  gemm_nt_wmma<<<gSmall, 256, 0, stream>>>(xbf, wGin, gin_b, tmpGin, IN_DIM, M_DIM);
  gate_combine<<<2048, 256, 0, stream>>>(tmpPre, tmpGin, gated, (long)T_DIM * M_DIM);

  // ---- 3) gout / skip GEMMs (logits stored; sigmoid fused into epilogue) --
  dim3 gBig(T_DIM / 64, OUT_DIM / 128);       // (128, 8)
  gemm_nt_wmma<<<gBig, 256, 0, stream>>>(xbf, wGout, gout_b, goutLog, IN_DIM, OUT_DIM);
  gemm_nt_wmma<<<gBig, 256, 0, stream>>>(xbf, wSkip, skip_b, skipBuf, IN_DIM, OUT_DIM);

  // ---- 4) chunked parallel scan -------------------------------------------
  dim3 gScan(NCH, M_DIM);                     // 4096 single-wave blocks
  scan_phase1<<<gScan, 32, 0, stream>>>(gated, done, a, b, bsumRe, bsumIm, anyd);
  scan_phase2<<<16, 256, 0, stream>>>(bsumRe, bsumIm, anyd, state_re, state_im,
                                      a, b, carryRe, carryIm);
  scan_phase3<<<gScan, 32, 0, stream>>>(gated, done, a, b, carryRe, carryIm,
                                        statesOut, zbf);

  // ---- 5) mix GEMM: [T, 2MC] bf16 x [OUT, 2MC] bf16 -> z fp32 -------------
  gemm_nt_wmma<<<gBig, 256, 0, stream>>>(zbf, wMix, mix_b, zBuf,
                                         2 * M_DIM * C_DIM, OUT_DIM);

  // ---- 6) fused sigmoid / LayerNorm / skip epilogue -----------------------
  ln_combine<<<T_DIM, 256, 0, stream>>>(zBuf, goutLog, skipBuf, outMain);
}
